// SelfAttentionPytorch_33569464385998
// MI455X (gfx1250) — matrix operand order repaired and checked by reading, not compiled, not run on hardware
//
#include <hip/hip_runtime.h>
#include <hip/hip_bf16.h>

#define BATCH 128
#define NTOK  64
#define DIMX  512
#define HEADS 8
#define HDIM  64   // DIMX / HEADS

typedef __bf16 bf16;
typedef __attribute__((ext_vector_type(16))) __bf16 v16bf;
typedef __attribute__((ext_vector_type(8)))  float  v8f;

union FragBF { v16bf v; unsigned int u[8]; };

// Async global->LDS path (gfx1250 ASYNCcnt), guarded so either toolchain compiles.
// Probe round revealed param type: pointer to int __attribute__((vector_size(16))).
#if __has_builtin(__builtin_amdgcn_global_load_async_to_lds_b128) && \
    __has_builtin(__builtin_amdgcn_s_wait_asynccnt)
#define USE_ASYNC_LDS 1
typedef int v4i __attribute__((vector_size(16)));
#define AS1(p) ((__attribute__((address_space(1))) v4i*)(p))
#define AS3(p) ((__attribute__((address_space(3))) v4i*)(p))
#else
#define USE_ASYNC_LDS 0
#endif

// ---------------------------------------------------------------------------
// WMMA wrapper: D = A(16x32 bf16) * B(32x16 bf16) + C(16x16 f32)
// ---------------------------------------------------------------------------
__device__ inline v8f wmma_bf16(v16bf a, v16bf b, v8f c) {
    return __builtin_amdgcn_wmma_f32_16x16x32_bf16(
        /*neg_a=*/false, a, /*neg_b=*/false, b,
        /*c_mod=*/(short)0, c, /*reuse_a=*/false, /*reuse_b=*/false);
}

// ---------------------------------------------------------------------------
// A-fragment (16x32 bf16, row-major in LDS: buf[row][col], ld = row stride).
// ISA layout: lane = {half(1b), m(4b)}; VGPR g holds K-pair at
//   col = col0 + 16*(g>>2) + 2*(g&3) + 8*half
// ---------------------------------------------------------------------------
__device__ inline v16bf frag_a(const bf16* buf, int ld, int row0, int col0, int lane) {
    FragBF f;
    const int half = lane >> 4;
    const int m    = lane & 15;
    const bf16* r = buf + (row0 + m) * ld + col0 + (half << 3);
#pragma unroll
    for (int g = 0; g < 8; ++g) {
        const int c = ((g >> 2) << 4) + ((g & 3) << 1);
        f.u[g] = *(const unsigned int*)(r + c);
    }
    return f.v;
}

// ---------------------------------------------------------------------------
// B-fragment (32x16 bf16). LDS holds B transposed: buf[n][k] (n = N column,
// k = K). ISA layout: lane = {half, n}; VGPR g holds K-pair at
//   k = col0 + 16*(g>>2) + 2*(g&3) + 8*half for column n0+n.
// ---------------------------------------------------------------------------
__device__ inline v16bf frag_b(const bf16* buf, int ld, int n0, int col0, int lane) {
    FragBF f;
    const int half = lane >> 4;
    const int n    = lane & 15;
    const bf16* r = buf + (n0 + n) * ld + col0 + (half << 3);
#pragma unroll
    for (int g = 0; g < 8; ++g)
        f.u[g] = *(const unsigned int*)(r + ((g >> 2) << 4) + ((g & 3) << 1));
    return f.v;
}

// ---------------------------------------------------------------------------
// Stage 1: q/k/v = (x @ W) as bf16, f32 accumulate.
// grid = (M/64, DIMX/64, 3), block = 256 (8 waves). Wave w owns output tiles
// (tm = w>>1, tn0 = (w&1)*2 .. +1) of the 64x64 block tile.
// ---------------------------------------------------------------------------
__global__ __launch_bounds__(256)
void qkv_proj_kernel(const float* __restrict__ x,
                     const float* __restrict__ Wq,
                     const float* __restrict__ Wk,
                     const float* __restrict__ Wv,
                     bf16* __restrict__ qkv_ws) {
    __shared__ __align__(16) bf16 XS[64][72];   // A tile, natural [m][k]
    __shared__ __align__(16) bf16 WST[64][72];  // B tile, transposed [n][k]

    const int gm  = blockIdx.x;      // 0..127 : 64-row slab of M = B*N
    const int gn  = blockIdx.y;      // 0..7   : 64-col slab of output
    const int wsl = blockIdx.z;      // 0..2   : which weight / output
    const float* W = (wsl == 0) ? Wq : (wsl == 1) ? Wk : Wv;
    bf16* Obuf = qkv_ws + (size_t)wsl * (size_t)(BATCH * NTOK) * DIMX;

    const int tid  = threadIdx.x;
    const int lane = tid & 31;
    const int wv   = tid >> 5;
    const int tm   = wv >> 1;            // 0..3
    const int tn0  = (wv & 1) << 1;      // 0 or 2

    const int lr = tid >> 2;             // 0..63
    const int lc = (tid & 3) << 4;       // 0,16,32,48

    v8f acc0 = {}; v8f acc1 = {};

    for (int kk = 0; kk < DIMX; kk += 64) {
        // --- X tile: 64x64 f32 -> bf16, natural layout ---
        {
            const float* src = x + (size_t)(gm * 64 + lr) * DIMX + kk + lc;
            const float4* s4 = (const float4*)src;
#pragma unroll
            for (int q = 0; q < 4; ++q) {
                float4 t = s4[q];
                XS[lr][lc + q * 4 + 0] = (bf16)t.x;
                XS[lr][lc + q * 4 + 1] = (bf16)t.y;
                XS[lr][lc + q * 4 + 2] = (bf16)t.z;
                XS[lr][lc + q * 4 + 3] = (bf16)t.w;
            }
            if (kk + 64 < DIMX) __builtin_prefetch(src + 64, 0, 3);
        }
        // --- W tile: 64x64 f32 -> bf16, transposed: WST[n][k] ---
        {
            const float* src = W + (size_t)(kk + lr) * DIMX + gn * 64 + lc;
            const float4* s4 = (const float4*)src;
#pragma unroll
            for (int q = 0; q < 4; ++q) {
                float4 t = s4[q];
                WST[lc + q * 4 + 0][lr] = (bf16)t.x;
                WST[lc + q * 4 + 1][lr] = (bf16)t.y;
                WST[lc + q * 4 + 2][lr] = (bf16)t.z;
                WST[lc + q * 4 + 3][lr] = (bf16)t.w;
            }
        }
        __syncthreads();
#pragma unroll
        for (int k2 = 0; k2 < 64; k2 += 32) {
            v16bf a  = frag_a(&XS[0][0], 72, tm * 16, k2, lane);
            v16bf b0 = frag_b(&WST[0][0], 72, tn0 * 16, k2, lane);
            v16bf b1 = frag_b(&WST[0][0], 72, (tn0 + 1) * 16, k2, lane);
            acc0 = wmma_bf16(a, b0, acc0);
            acc1 = wmma_bf16(a, b1, acc1);
        }
        __syncthreads();
    }

    // store two 16x16 f32 tiles as bf16 (C/D layout: VGPR g -> row g + 8*half)
    const int half = lane >> 4;
    const int cn   = lane & 15;
#pragma unroll
    for (int g = 0; g < 8; ++g) {
        const int m = tm * 16 + g + (half << 3);
        const size_t row = (size_t)(gm * 64 + m);
        Obuf[row * DIMX + gn * 64 + tn0 * 16 + cn]       = (bf16)acc0[g];
        Obuf[row * DIMX + gn * 64 + (tn0 + 1) * 16 + cn] = (bf16)acc1[g];
    }
}

// ---------------------------------------------------------------------------
// Stage 2: per (batch, head): S = QK^T/8, P = softmax_rows(S), O = P V,
// out = O + x. One block per (b,h): 1024 blocks of 256 threads.
// ---------------------------------------------------------------------------
__global__ __launch_bounds__(256)
void attention_kernel(const float* __restrict__ x,
                      const bf16* __restrict__ qkv_ws,
                      float* __restrict__ out) {
    __shared__ __align__(16) bf16  QS[64][72];   // [n][d]
    __shared__ __align__(16) bf16  KS[64][72];   // [m][d]  (== B^T for S)
    __shared__ __align__(16) bf16  VST[64][72];  // [d][m]  (== B^T for O)
    __shared__ __align__(16) float SS[64][68];   // scores f32
    __shared__ __align__(16) bf16  PS[64][72];   // softmax probs bf16

    const int b = blockIdx.x >> 3;   // 0..127
    const int h = blockIdx.x & 7;    // 0..7
    const bf16* qg = qkv_ws;
    const bf16* kg = qg + (size_t)(BATCH * NTOK) * DIMX;
    const bf16* vg = kg + (size_t)(BATCH * NTOK) * DIMX;

    const int tid  = threadIdx.x;
    const int lane = tid & 31;
    const int wv   = tid >> 5;
    const int tm   = wv >> 1;
    const int tn0  = (wv & 1) << 1;

    const int lr = tid >> 2;             // 0..63
    const int lc = (tid & 3) << 4;       // 0,16,32,48

    // --- load head slices: Q,K natural (async if available), V transposed ---
    {
        const size_t base = (size_t)(b * NTOK + lr) * DIMX + h * HDIM + lc;
#if USE_ASYNC_LDS
        // bf16 pure copies: stream straight into LDS, tracked by ASYNCcnt.
        __builtin_amdgcn_global_load_async_to_lds_b128(AS1(qg + base),     AS3(&QS[lr][lc]),     0, 0);
        __builtin_amdgcn_global_load_async_to_lds_b128(AS1(qg + base + 8), AS3(&QS[lr][lc + 8]), 0, 0);
        __builtin_amdgcn_global_load_async_to_lds_b128(AS1(kg + base),     AS3(&KS[lr][lc]),     0, 0);
        __builtin_amdgcn_global_load_async_to_lds_b128(AS1(kg + base + 8), AS3(&KS[lr][lc + 8]), 0, 0);
#else
        const uint4* qs = (const uint4*)(qg + base);
        uint4* qd = (uint4*)&QS[lr][lc];
        qd[0] = qs[0]; qd[1] = qs[1];
        const uint4* ks = (const uint4*)(kg + base);
        uint4* kd = (uint4*)&KS[lr][lc];
        kd[0] = ks[0]; kd[1] = ks[1];
#endif
        // V needs a transpose for the P@V B-operand: via VGPRs, overlaps async.
        bf16 tmp[16];
        *(uint4*)&tmp[0] = ((const uint4*)(vg + base))[0];
        *(uint4*)&tmp[8] = ((const uint4*)(vg + base))[1];
#pragma unroll
        for (int j = 0; j < 16; ++j) VST[lc + j][lr] = tmp[j];  // VST[d][m]
#if USE_ASYNC_LDS
        __builtin_amdgcn_s_wait_asynccnt(0);
#endif
    }
    __syncthreads();

    // --- S = Q @ K^T, scaled by 1/sqrt(64) = 1/8 ---
    {
        v8f acc0 = {}; v8f acc1 = {};
#pragma unroll
        for (int k2 = 0; k2 < 64; k2 += 32) {
            v16bf a  = frag_a(&QS[0][0], 72, tm * 16, k2, lane);
            v16bf b0 = frag_b(&KS[0][0], 72, tn0 * 16, k2, lane);
            v16bf b1 = frag_b(&KS[0][0], 72, (tn0 + 1) * 16, k2, lane);
            acc0 = wmma_bf16(a, b0, acc0);
            acc1 = wmma_bf16(a, b1, acc1);
        }
        const int half = lane >> 4;
        const int cn   = lane & 15;
#pragma unroll
        for (int g = 0; g < 8; ++g) {
            const int m = tm * 16 + g + (half << 3);
            SS[m][tn0 * 16 + cn]       = acc0[g] * 0.125f;
            SS[m][(tn0 + 1) * 16 + cn] = acc1[g] * 0.125f;
        }
    }
    __syncthreads();

    // --- row softmax over keys; write bf16 probabilities ---
    if (tid < 64) {
        float mx = -3.402823466e38f;
        for (int j = 0; j < 64; ++j) mx = fmaxf(mx, SS[tid][j]);
        float sum = 0.f;
        for (int j = 0; j < 64; ++j) {
            const float e = __expf(SS[tid][j] - mx);
            SS[tid][j] = e;
            sum += e;
        }
        const float inv = 1.f / sum;
        for (int j = 0; j < 64; ++j) PS[tid][j] = (bf16)(SS[tid][j] * inv);
    }
    __syncthreads();

    // --- O = P @ V ; out = O + x (residual) ---
    {
        v8f acc0 = {}; v8f acc1 = {};
#pragma unroll
        for (int k2 = 0; k2 < 64; k2 += 32) {
            v16bf a  = frag_a(&PS[0][0], 72, tm * 16, k2, lane);
            v16bf b0 = frag_b(&VST[0][0], 72, tn0 * 16, k2, lane);
            v16bf b1 = frag_b(&VST[0][0], 72, (tn0 + 1) * 16, k2, lane);
            acc0 = wmma_bf16(a, b0, acc0);
            acc1 = wmma_bf16(a, b1, acc1);
        }
        const int half = lane >> 4;
        const int cn   = lane & 15;
#pragma unroll
        for (int g = 0; g < 8; ++g) {
            const int m = tm * 16 + g + (half << 3);
            const size_t o = (size_t)(b * NTOK + m) * DIMX + h * HDIM;
            out[o + tn0 * 16 + cn]       = acc0[g] + x[o + tn0 * 16 + cn];
            out[o + (tn0 + 1) * 16 + cn] = acc1[g] + x[o + (tn0 + 1) * 16 + cn];
        }
    }
}

// ---------------------------------------------------------------------------
extern "C" void kernel_launch(void* const* d_in, const int* in_sizes, int n_in,
                              void* d_out, int out_size, void* d_ws, size_t ws_size,
                              hipStream_t stream) {
    const float* x = (const float*)d_in[0];
    const float* Q = (const float*)d_in[1];
    const float* K = (const float*)d_in[2];
    const float* V = (const float*)d_in[3];
    // d_in[4] = number (scalar, fixed to 64) — unused
    float* out = (float*)d_out;
    bf16* qkv = (bf16*)d_ws;   // 3 * 8192*512 bf16 = 24 MB scratch

    dim3 g1(BATCH * NTOK / 64, DIMX / 64, 3);
    qkv_proj_kernel<<<g1, 256, 0, stream>>>(x, Q, K, V, qkv);

    dim3 g2(BATCH * HEADS);
    attention_kernel<<<g2, 256, 0, stream>>>(x, qkv, out);
}
